// SingleSAGPool_2834678415644
// MI455X (gfx1250) — compile-verified, matrix operand order checked by reading, not run on hardware
//
#include <hip/hip_runtime.h>
#include <math.h>

// ---------------------------------------------------------------------------
// Problem constants (match reference)
// ---------------------------------------------------------------------------
#define BGRAPHS 64
#define NPG     2048
#define DEG     8
#define NEDGE   (BGRAPHS * NPG * DEG)   // 1,048,576

typedef __attribute__((ext_vector_type(16))) _Float16 v16h;
typedef __attribute__((ext_vector_type(8)))  _Float16 v8h;
typedef __attribute__((ext_vector_type(8)))  float    v8f;

static __device__ __forceinline__ unsigned fkey(float f) {
    unsigned u = __float_as_uint(f);
    return (u & 0x80000000u) ? ~u : (u | 0x80000000u);
}
static __device__ __forceinline__ float funkey(unsigned k) {
    unsigned u = (k & 0x80000000u) ? (k & 0x7fffffffu) : ~k;
    return __uint_as_float(u);
}

// Fragment = 16 halfs per lane: elements 0..7 at p[0..7], 8..15 at p[16..23]
// (matches the 16-bit A/B VGPR layout: K = half*8 + {0..7, 16..23})
static __device__ __forceinline__ v16h ld_frag(const _Float16* __restrict__ p) {
    v8h lo = *(const v8h*)p;
    v8h hi = *(const v8h*)(p + 16);
    return __builtin_shufflevector(lo, hi, 0, 1, 2, 3, 4, 5, 6, 7,
                                           8, 9, 10, 11, 12, 13, 14, 15);
}

// ---------------------------------------------------------------------------
// f32 -> f16 streaming convert (row-major layout preserved)
// ---------------------------------------------------------------------------
__global__ void cvt_f16(const float* __restrict__ in, _Float16* __restrict__ out, int n)
{
    int i = blockIdx.x * blockDim.x + threadIdx.x;
    if (i < n) out[i] = (_Float16)in[i];
}

// ---------------------------------------------------------------------------
// Repack W:[fin,hc] f32 -> Wp[t][n][kk] f16 with t=K-tile(32), n=col, kk=0..31
// so a lane's B fragment (col n, K pattern) is two contiguous b128 loads.
// ---------------------------------------------------------------------------
__global__ void pack_w(const float* __restrict__ W, _Float16* __restrict__ Wp,
                       int fin, int hc)
{
    int idx = blockIdx.x * blockDim.x + threadIdx.x;   // over (fin/32)*hc
    int tot = (fin >> 5) * hc;
    if (idx >= tot) return;
    int t = idx / hc, n = idx - t * hc;
    const float* src = W + (size_t)(t * 32) * hc + n;
    _Float16*    dst = Wp + (size_t)idx * 32;
#pragma unroll
    for (int kk = 0; kk < 32; ++kk) dst[kk] = (_Float16)src[(size_t)kk * hc];
}

// ---------------------------------------------------------------------------
// Dual GEMM via WMMA: xl = h@Wl + bl ; xr = h@Wr + br
// One wave per 16(M) x 32(N) tile: 4 wmma per K-step, A fragment reused.
// ---------------------------------------------------------------------------
__global__ __launch_bounds__(32) void gemm_dual_wmma(
    const _Float16* __restrict__ hH,
    const _Float16* __restrict__ WlP, const float* __restrict__ bl,
    const _Float16* __restrict__ WrP, const float* __restrict__ br,
    float* __restrict__ xl, float* __restrict__ xr,
    int M, int fin, int hc)
{
    const int ncg  = hc >> 5;                 // 32-wide column groups
    const int m0   = (blockIdx.x / ncg) << 4;
    const int n0   = (blockIdx.x % ncg) << 5;
    const int lane = threadIdx.x & 31;
    const int lrow = lane & 15;
    const int half = lane >> 4;

    v8f cl0, cl1, cr0, cr1;
    const float bv0 = bl[n0 + lrow],      bv1 = bl[n0 + 16 + lrow];
    const float cv0 = br[n0 + lrow],      cv1 = br[n0 + 16 + lrow];
#pragma unroll
    for (int r = 0; r < 8; ++r) { cl0[r] = bv0; cl1[r] = bv1; cr0[r] = cv0; cr1[r] = cv1; }

    const _Float16* ap = hH  + (size_t)(m0 + lrow) * fin + half * 8;
    const _Float16* lp = WlP + (size_t)(n0 + lrow) * 32 + half * 8;
    const _Float16* rp = WrP + (size_t)(n0 + lrow) * 32 + half * 8;

    for (int k0 = 0; k0 < fin; k0 += 32) {
        const size_t wof = (size_t)(k0 >> 5) * hc * 32;
        v16h a   = ld_frag(ap + k0);
        v16h fl0 = ld_frag(lp + wof);
        v16h fl1 = ld_frag(lp + wof + 16 * 32);
        v16h fr0 = ld_frag(rp + wof);
        v16h fr1 = ld_frag(rp + wof + 16 * 32);
        __builtin_prefetch(ap + k0 + 32, 0, 1);
        cl0 = __builtin_amdgcn_wmma_f32_16x16x32_f16(false, a, false, fl0,
                                                     (short)0, cl0, false, false);
        cl1 = __builtin_amdgcn_wmma_f32_16x16x32_f16(false, a, false, fl1,
                                                     (short)0, cl1, false, false);
        cr0 = __builtin_amdgcn_wmma_f32_16x16x32_f16(false, a, false, fr0,
                                                     (short)0, cr0, false, false);
        cr1 = __builtin_amdgcn_wmma_f32_16x16x32_f16(false, a, false, fr1,
                                                     (short)0, cr1, false, false);
    }
#pragma unroll
    for (int r = 0; r < 8; ++r) {
        const int row = m0 + r + half * 8;
        xl[(size_t)row * hc + n0 + lrow]      = cl0[r];
        xl[(size_t)row * hc + n0 + 16 + lrow] = cl1[r];
        xr[(size_t)row * hc + n0 + lrow]      = cr0[r];
        xr[(size_t)row * hc + n0 + 16 + lrow] = cr1[r];
    }
}

// ---------------------------------------------------------------------------
// Scorer transform: o[i] = dot(h[i,:], w) + b  (hc -> 1)
// ---------------------------------------------------------------------------
__global__ void vec_transform(const float* __restrict__ h, const float* __restrict__ w,
                              const float* __restrict__ b, float* __restrict__ o,
                              int N, int hc)
{
    int i = blockIdx.x * blockDim.x + threadIdx.x;
    if (i >= N) return;
    float acc = b[0];
    const float* hr = h + (size_t)i * hc;
    for (int c = 0; c < hc; ++c) acc += hr[c] * w[c];
    o[i] = acc;
}

// ---------------------------------------------------------------------------
// Edge pass 1: per (edge, head) logit + segment max (ordered-uint atomicMax)
// Edges e >= E are self loops (s = d = e - E), always unmasked.
// ---------------------------------------------------------------------------
__global__ void edge_logits(const int* __restrict__ src, const int* __restrict__ dst,
                            const int* __restrict__ emask,
                            const float* __restrict__ xl, const float* __restrict__ xr,
                            const float* __restrict__ att,
                            float* __restrict__ logits, unsigned* __restrict__ mxkey,
                            int E, int Etot, int H, int C)
{
    int idx = blockIdx.x * blockDim.x + threadIdx.x;
    if (idx >= Etot * H) return;
    int e = idx / H, hh = idx - e * H;
    int s, d, m;
    if (e < E) { m = emask[e]; s = src[e]; d = dst[e]; }
    else       { m = 1; s = d = e - E; }
    if (!m) { logits[idx] = -1e9f; return; }
    const int hc = H * C;
    const float* pl = xl + (size_t)s * hc + hh * C;
    const float* pr = xr + (size_t)d * hc + hh * C;
    const float* pa = att + hh * C;
    float acc = 0.f;
    for (int c = 0; c < C; ++c) {
        float v = pl[c] + pr[c];
        v = v > 0.f ? v : 0.2f * v;          // leaky_relu(0.2)
        acc += v * pa[c];
    }
    logits[idx] = acc;
    atomicMax(mxkey + (size_t)d * H + hh, fkey(acc));
}

// ---------------------------------------------------------------------------
// Edge pass 2: ex = exp(logit - max[d]); den[d] += ex
// ---------------------------------------------------------------------------
__global__ void edge_exp(const int* __restrict__ dst, const int* __restrict__ emask,
                         const unsigned* __restrict__ mxkey,
                         float* __restrict__ logits, float* __restrict__ den,
                         int E, int Etot, int H)
{
    int idx = blockIdx.x * blockDim.x + threadIdx.x;
    if (idx >= Etot * H) return;
    int e = idx / H, hh = idx - e * H;
    int d, m;
    if (e < E) { m = emask[e]; d = dst[e]; }
    else       { m = 1; d = e - E; }
    if (!m) { logits[idx] = 0.f; return; }
    float mx = funkey(mxkey[(size_t)d * H + hh]);
    float ex = __expf(logits[idx] - mx);
    logits[idx] = ex;
    atomicAdd(den + (size_t)d * H + hh, ex);
}

// ---------------------------------------------------------------------------
// Edge pass 3: out[d,:] += xl[s,:] * alpha
// ---------------------------------------------------------------------------
__global__ void edge_aggr(const int* __restrict__ src, const int* __restrict__ dst,
                          const int* __restrict__ emask,
                          const float* __restrict__ xl, const float* __restrict__ ex,
                          const float* __restrict__ den, float* __restrict__ out,
                          int E, int Etot, int H, int C)
{
    int idx = blockIdx.x * blockDim.x + threadIdx.x;
    if (idx >= Etot * H) return;
    int e = idx / H, hh = idx - e * H;
    int s, d, m;
    if (e < E) { m = emask[e]; s = src[e]; d = dst[e]; }
    else       { m = 1; s = d = e - E; }
    if (!m) return;
    float alpha = ex[idx] / (den[(size_t)d * H + hh] + 1e-16f);
    const int hc = H * C;
    const float* pl = xl + (size_t)s * hc + hh * C;
    float*       po = out + (size_t)d * hc + hh * C;
    for (int c = 0; c < C; ++c) atomicAdd(po + c, pl[c] * alpha);
}

__global__ void bias_relu(float* __restrict__ h, const float* __restrict__ bias,
                          int N, int hc)
{
    int idx = blockIdx.x * blockDim.x + threadIdx.x;
    if (idx >= N * hc) return;
    float v = h[idx] + bias[idx % hc];
    h[idx] = v > 0.f ? v : 0.f;
}

__global__ void tanh_score(const float* __restrict__ acc, const float* __restrict__ pbias,
                           float* __restrict__ score, int N)
{
    int i = blockIdx.x * blockDim.x + threadIdx.x;
    if (i >= N) return;
    score[i] = tanhf(acc[i] + pbias[0]);
}

// ---------------------------------------------------------------------------
// Per-graph top-k via stable ranking (matches stable argsort(-score)).
// ---------------------------------------------------------------------------
__global__ __launch_bounds__(256) void topk_kernel(const float* __restrict__ score,
                                                   int* __restrict__ newid,
                                                   int* __restrict__ perm,
                                                   int n, int k)
{
    __shared__ float s[2048];
    const int g = blockIdx.x;
    const float* sg = score + (size_t)g * n;
    for (int i = threadIdx.x; i < n; i += blockDim.x) s[i] = sg[i];
    __syncthreads();
    for (int i = threadIdx.x; i < n; i += blockDim.x) {
        const float si = s[i];
        int rank = 0;
        for (int j = 0; j < n; ++j) {
            const float sj = s[j];
            rank += (sj > si) || (sj == si && j < i);
        }
        if (rank < k) { newid[g * n + i] = g * k + rank; perm[g * k + rank] = g * n + i; }
        else          { newid[g * n + i] = -1; }
    }
}

__global__ void gather_gate(const float* __restrict__ h, const float* __restrict__ score,
                            const int* __restrict__ perm, float* __restrict__ hnew,
                            int Nnew, int hc)
{
    int idx = blockIdx.x * blockDim.x + threadIdx.x;
    if (idx >= Nnew * hc) return;
    int j = idx / hc, c = idx - j * hc;
    int old = perm[j];
    hnew[idx] = h[(size_t)old * hc + c] * score[old];
}

__global__ void edge_remap(int* __restrict__ src, int* __restrict__ dst,
                           int* __restrict__ emask, const int* __restrict__ newid, int E)
{
    int e = blockIdx.x * blockDim.x + threadIdx.x;
    if (e >= E) return;
    if (!emask[e]) { src[e] = 0; dst[e] = 0; return; }
    int ns = newid[src[e]], nd = newid[dst[e]];
    int ok = (ns >= 0) && (nd >= 0);
    emask[e] = ok;
    src[e] = ok ? ns : 0;
    dst[e] = ok ? nd : 0;
}

__global__ void init_edges(const int* __restrict__ ei, int* __restrict__ src,
                           int* __restrict__ dst, int* __restrict__ emask, int E)
{
    int e = blockIdx.x * blockDim.x + threadIdx.x;
    if (e >= E) return;
    src[e] = ei[e];
    dst[e] = ei[E + e];
    emask[e] = 1;
}

// Mean pool over k nodes + 2-way prediction head. One block per graph.
__global__ __launch_bounds__(256) void pool_pred(const float* __restrict__ h,
                                                 const float* __restrict__ predW,
                                                 const float* __restrict__ predb,
                                                 float* __restrict__ out, int k, int hc)
{
    __shared__ float pooled[128];
    const int g = blockIdx.x;
    for (int c = threadIdx.x; c < hc; c += blockDim.x) {
        float acc = 0.f;
        const float* base = h + (size_t)g * k * hc + c;
        for (int j = 0; j < k; ++j) acc += base[(size_t)j * hc];
        pooled[c] = acc / (float)k;
    }
    __syncthreads();
    if (threadIdx.x < 2) {
        float acc = predb[threadIdx.x];
        for (int c = 0; c < hc; ++c) acc += pooled[c] * predW[c * 2 + threadIdx.x];
        out[g * 2 + threadIdx.x] = acc;
    }
}

// Final: mean pool (k nodes, 32 ch) -> relu(lin1) -> lin2. One block per graph.
__global__ __launch_bounds__(64) void final_head(const float* __restrict__ h,
                                                 const float* __restrict__ W1,
                                                 const float* __restrict__ b1,
                                                 const float* __restrict__ W2,
                                                 const float* __restrict__ b2,
                                                 float* __restrict__ out, int k)
{
    __shared__ float pooled[32];
    __shared__ float z[32];
    const int g = blockIdx.x;
    const int hc = 32;
    for (int c = threadIdx.x; c < hc; c += blockDim.x) {
        float acc = 0.f;
        const float* base = h + (size_t)g * k * hc + c;
        for (int j = 0; j < k; ++j) acc += base[(size_t)j * hc];
        pooled[c] = acc / (float)k;
    }
    __syncthreads();
    if (threadIdx.x < 32) {
        float acc = b1[threadIdx.x];
        for (int c = 0; c < 32; ++c) acc += pooled[c] * W1[c * 32 + threadIdx.x];
        z[threadIdx.x] = acc > 0.f ? acc : 0.f;
    }
    __syncthreads();
    if (threadIdx.x < 2) {
        float acc = b2[threadIdx.x];
        for (int c = 0; c < 32; ++c) acc += z[c] * W2[c * 2 + threadIdx.x];
        out[g * 2 + threadIdx.x] = acc;
    }
}

// ---------------------------------------------------------------------------
// Host orchestration
// ---------------------------------------------------------------------------
static inline int cdiv(long long a, int b) { return (int)((a + b - 1) / b); }

extern "C" void kernel_launch(void* const* d_in, const int* in_sizes, int n_in,
                              void* d_out, int out_size, void* d_ws, size_t ws_size,
                              hipStream_t stream)
{
    (void)in_sizes; (void)n_in; (void)out_size; (void)ws_size;

    const int NL[4]  = {2048, 1024, 512, 256};
    const int KL[3]  = {1024, 512, 256};
    const int FIN[3] = {64, 128, 128};
    const int HH[3]  = {4, 4, 1};
    const int CC[3]  = {32, 32, 32};
    const int E = NEDGE;

    // d_in layout (setup_inputs dict insertion order, leaves flattened):
    // 0: x ; 1 + 14*i + {0..13}: Wl,bl,Wr,br,att,bias,pWl,pbl,pWr,pbr,patt,pbias,predW,predb
    // 43..46: lin1W, lin1b, lin2W, lin2b ; 47: edge_index ; 48: batch (unused)
    auto F = [&](int i) -> const float* { return (const float*)d_in[i]; };
    const float* x = F(0);
    const int* edge_index = (const int*)d_in[47];
    float* outF = (float*)d_out;

    // Workspace carve-up
    char* w = (char*)d_ws;
    auto alloc = [&](size_t bytes) -> void* {
        void* p = (void*)w;
        w += (bytes + 255) & ~(size_t)255;
        return p;
    };
    const size_t maxNhc = (size_t)BGRAPHS * NPG * 128;       // 16.7M floats
    float*     xl      = (float*)alloc(maxNhc * 4);
    float*     xr      = (float*)alloc(maxNhc * 4);
    float*     hConv   = (float*)alloc(maxNhc * 4);
    float*     hPool   = (float*)alloc((size_t)BGRAPHS * 1024 * 128 * 4);
    _Float16*  hHalf   = (_Float16*)alloc((size_t)BGRAPHS * NPG * 128 * 2);
    _Float16*  WlP     = (_Float16*)alloc((size_t)128 * 128 * 2);
    _Float16*  WrP     = (_Float16*)alloc((size_t)128 * 128 * 2);
    float*     edgebuf = (float*)alloc((size_t)(E + BGRAPHS * NPG) * 4 * 4);
    unsigned*  mxkey   = (unsigned*)alloc((size_t)BGRAPHS * NPG * 4 * 4);
    float*     den     = (float*)alloc((size_t)BGRAPHS * NPG * 4 * 4);
    float*     xlp     = (float*)alloc((size_t)BGRAPHS * NPG * 4);
    float*     xrp     = (float*)alloc((size_t)BGRAPHS * NPG * 4);
    float*     sacc    = (float*)alloc((size_t)BGRAPHS * NPG * 4);
    float*     score   = (float*)alloc((size_t)BGRAPHS * NPG * 4);
    int*       newid   = (int*)alloc((size_t)BGRAPHS * NPG * 4);
    int*       perm    = (int*)alloc((size_t)BGRAPHS * 1024 * 4);
    int*       srcB    = (int*)alloc((size_t)E * 4);
    int*       dstB    = (int*)alloc((size_t)E * 4);
    int*       emask   = (int*)alloc((size_t)E * 4);

    init_edges<<<cdiv(E, 256), 256, 0, stream>>>(edge_index, srcB, dstB, emask, E);

    const float* hin = x;
    for (int i = 0; i < 3; ++i) {
        const int base = 1 + 14 * i;
        const float* Wl    = F(base + 0);
        const float* bl    = F(base + 1);
        const float* Wr    = F(base + 2);
        const float* br    = F(base + 3);
        const float* att   = F(base + 4);
        const float* bias  = F(base + 5);
        const float* pWl   = F(base + 6);
        const float* pbl   = F(base + 7);
        const float* pWr   = F(base + 8);
        const float* pbr   = F(base + 9);
        const float* patt  = F(base + 10);
        const float* pbias = F(base + 11);
        const float* predW = F(base + 12);
        const float* predb = F(base + 13);

        const int Ncur = BGRAPHS * NL[i];
        const int fin  = FIN[i];
        const int H    = HH[i];
        const int C    = CC[i];
        const int hc   = H * C;
        const int Etot = E + Ncur;
        const int nEH  = Etot * H;

        // ---- conv GATv2: dual WMMA GEMM with f16 staging ----
        hipMemsetAsync(hConv, 0, (size_t)Ncur * hc * 4, stream);
        hipMemsetAsync(mxkey, 0, (size_t)Ncur * H * 4, stream);
        hipMemsetAsync(den,   0, (size_t)Ncur * H * 4, stream);
        cvt_f16<<<cdiv((long long)Ncur * fin, 256), 256, 0, stream>>>(
            hin, hHalf, Ncur * fin);
        pack_w<<<cdiv((fin >> 5) * hc, 128), 128, 0, stream>>>(Wl, WlP, fin, hc);
        pack_w<<<cdiv((fin >> 5) * hc, 128), 128, 0, stream>>>(Wr, WrP, fin, hc);
        gemm_dual_wmma<<<(Ncur / 16) * (hc / 32), 32, 0, stream>>>(
            hHalf, WlP, bl, WrP, br, xl, xr, Ncur, fin, hc);
        edge_logits<<<cdiv(nEH, 256), 256, 0, stream>>>(
            srcB, dstB, emask, xl, xr, att, edgebuf, mxkey, E, Etot, H, C);
        edge_exp<<<cdiv(nEH, 256), 256, 0, stream>>>(
            dstB, emask, mxkey, edgebuf, den, E, Etot, H);
        edge_aggr<<<cdiv(nEH, 256), 256, 0, stream>>>(
            srcB, dstB, emask, xl, edgebuf, den, hConv, E, Etot, H, C);
        bias_relu<<<cdiv((long long)Ncur * hc, 256), 256, 0, stream>>>(
            hConv, bias, Ncur, hc);

        // ---- SAGPool scorer GATv2 (H=1, C=1) ----
        hipMemsetAsync(sacc,  0, (size_t)Ncur * 4, stream);
        hipMemsetAsync(mxkey, 0, (size_t)Ncur * 4, stream);
        hipMemsetAsync(den,   0, (size_t)Ncur * 4, stream);
        vec_transform<<<cdiv(Ncur, 256), 256, 0, stream>>>(hConv, pWl, pbl, xlp, Ncur, hc);
        vec_transform<<<cdiv(Ncur, 256), 256, 0, stream>>>(hConv, pWr, pbr, xrp, Ncur, hc);
        edge_logits<<<cdiv(Etot, 256), 256, 0, stream>>>(
            srcB, dstB, emask, xlp, xrp, patt, edgebuf, mxkey, E, Etot, 1, 1);
        edge_exp<<<cdiv(Etot, 256), 256, 0, stream>>>(
            dstB, emask, mxkey, edgebuf, den, E, Etot, 1);
        edge_aggr<<<cdiv(Etot, 256), 256, 0, stream>>>(
            srcB, dstB, emask, xlp, edgebuf, den, sacc, E, Etot, 1, 1);
        tanh_score<<<cdiv(Ncur, 256), 256, 0, stream>>>(sacc, pbias, score, Ncur);

        // ---- top-k, gather+gate, pred head, edge remap ----
        topk_kernel<<<BGRAPHS, 256, 0, stream>>>(score, newid, perm, NL[i], KL[i]);
        const int Nnew = BGRAPHS * KL[i];
        gather_gate<<<cdiv((long long)Nnew * hc, 256), 256, 0, stream>>>(
            hConv, score, perm, hPool, Nnew, hc);
        pool_pred<<<BGRAPHS, 256, 0, stream>>>(
            hPool, predW, predb, outF + (1 + i) * BGRAPHS * 2, KL[i], hc);
        edge_remap<<<cdiv(E, 256), 256, 0, stream>>>(srcB, dstB, emask, newid, E);

        hin = hPool;
    }

    final_head<<<BGRAPHS, 64, 0, stream>>>(
        hPool, F(43), F(44), F(45), F(46), outF, KL[2]);
}
